// MultiHeadedAttention_88210038325999
// MI455X (gfx1250) — matrix operand order repaired and checked by reading, not compiled, not run on hardware
//
#include <hip/hip_runtime.h>
#include <hip/hip_bf16.h>

typedef __attribute__((ext_vector_type(16))) _Float16 v16h;
typedef __attribute__((ext_vector_type(8)))  _Float16 v8h;
typedef __attribute__((ext_vector_type(8)))  float    v8f;

union H16 { v16h v; struct { v8h lo; v8h hi; } s; };

__device__ __forceinline__ v8f wmma_f16(v16h a, v16h b, v8f c) {
  return __builtin_amdgcn_wmma_f32_16x16x32_f16(false, a, false, b, (short)0, c, false, false);
}

// CDNA5 async DMA: 16B global -> LDS, tracked by ASYNCcnt (no VGPR data path).
__device__ __forceinline__ void async_copy_b128(unsigned lds_off, const _Float16* g) {
  asm volatile("global_load_async_to_lds_b128 %0, %1, off"
               :: "v"(lds_off), "v"(g) : "memory");
}
__device__ __forceinline__ void wait_async0() {
  asm volatile("s_wait_asynccnt 0" ::: "memory");
}

#define NSEQ 2048
#define DMODEL 256
#define NB 4
#define NH 4

// ---------------------------------------------------------------------------
// Kernel 1: QKV projection + bias + RoPE, f32 -> f16.
// Q,K in [b][h][n][64]; V stored TRANSPOSED as [b][h][d][n] so the attention
// kernel can stage both K and V chunks with pure async row copies.
// blockIdx: x = tile group (256), y = batch (4), z = which matrix (q/k/v)
// ---------------------------------------------------------------------------
__global__ __launch_bounds__(256) void qkv_proj_rope_kernel(
    const float* __restrict__ q_in, const float* __restrict__ k_in, const float* __restrict__ v_in,
    const float* __restrict__ Wq, const float* __restrict__ Wk, const float* __restrict__ Wv,
    const float* __restrict__ bq, const float* __restrict__ bk, const float* __restrict__ bv,
    const float* __restrict__ enc,
    _Float16* __restrict__ Qh, _Float16* __restrict__ Kh, _Float16* __restrict__ Vt)
{
  const int mat = blockIdx.z;
  const float* X; const float* W; const float* bias; _Float16* out;
  bool rope; float scale;
  if (mat == 0)      { X = q_in; W = Wq; bias = bq; out = Qh; rope = true;  scale = 0.125f; } // fold 1/sqrt(64)
  else if (mat == 1) { X = k_in; W = Wk; bias = bk; out = Kh; rope = true;  scale = 1.0f;   }
  else               { X = v_in; W = Wv; bias = bv; out = Vt; rope = false; scale = 1.0f;   }

  const int b    = blockIdx.y;
  const int lane = threadIdx.x & 31;
  const int wave = threadIdx.x >> 5;
  const int tile = blockIdx.x * 8 + wave;          // 0..2047
  const int o0   = (tile & 15) * 16;               // output-channel tile
  const int n0   = (tile >> 4) * 16;               // sequence tile
  const int half = lane >> 4;
  const int lan  = lane & 15;
  const int n    = n0 + lan;

  const float* Xb = X + (size_t)b * (DMODEL * NSEQ);
  const float* wr = W + (size_t)(o0 + lan) * DMODEL;   // A row for this lane

  v8f acc = {};
  #pragma unroll
  for (int k0 = 0; k0 < DMODEL; k0 += 32) {
    v16h a, bb;
    #pragma unroll
    for (int j = 0; j < 8; ++j) {                  // A frag: K = half*16 + {0..15}
      a[j]     = (_Float16)wr[k0 + half * 16 + j];
      a[j + 8] = (_Float16)wr[k0 + half * 16 + 8 + j];
    }
    #pragma unroll
    for (int j = 0; j < 16; ++j)                   // B frag: column n, K = half*16 + j
      bb[j] = (_Float16)Xb[(size_t)(k0 + half * 16 + j) * NSEQ + n];
    acc = wmma_f16(a, bb, acc);
  }
  #pragma unroll
  for (int r = 0; r < 8; ++r) acc[r] += bias[o0 + half * 8 + r];

  if (rope) {
    // channel c = o0 + half*8 + r ; head dim dd = c>>2 ; pairs (r, r+4) = dims (ddA, ddA+1)
    const int ddA = (o0 >> 2) + half * 2;
    const float c0 = enc[(size_t)n * 64 + ddA];
    const float c1 = enc[(size_t)n * 64 + ddA + 1];
    const float s0 = enc[(size_t)(NSEQ + n) * 64 + ddA];
    const float s1 = enc[(size_t)(NSEQ + n) * 64 + ddA + 1];
    #pragma unroll
    for (int r = 0; r < 4; ++r) {
      float e = acc[r], o = acc[r + 4];
      acc[r]     = e * c0 - o * s0;
      acc[r + 4] = o * c1 + e * s1;
    }
  }
  #pragma unroll
  for (int r = 0; r < 8; ++r) {
    int c  = o0 + half * 8 + r;
    int hh = c & 3, dd = c >> 2;
    _Float16 val = (_Float16)(acc[r] * scale);
    if (mat == 2)   // V transposed: [b][h][d][n]
      out[((size_t)(b * NH + hh) * 64 + dd) * NSEQ + n] = val;
    else            // Q/K: [b][h][n][d]
      out[((size_t)(b * NH + hh) * NSEQ + n) * 64 + dd] = val;
  }
}

// ---------------------------------------------------------------------------
// Kernel 2: flash attention per (b,h). Block = 8 waves x 16 query rows = 128.
// K (32x64) and V^T (64x32) chunks staged with double-buffered
// global_load_async_to_lds_b128 DMA, synchronized via ASYNCcnt.
// blockIdx: x = query block (16), y = head (4), z = batch (4)
// Output Xh in [b][c=dd*4+h][n] f16 for the final projection.
// ---------------------------------------------------------------------------
__global__ __launch_bounds__(256) void attention_kernel(
    const _Float16* __restrict__ Qh, const _Float16* __restrict__ Kh,
    const _Float16* __restrict__ Vt, _Float16* __restrict__ Xh)
{
  __shared__ __align__(32) _Float16 sK[2][32 * 64];   // [buf][m_local][d]
  __shared__ __align__(32) _Float16 sV[2][64 * 32];   // [buf][d][m_local]
  __shared__ __align__(32) _Float16 sP[8][16 * 32];   // per-wave P transpose buffer

  const int b = blockIdx.z, h = blockIdx.y;
  const int lane = threadIdx.x & 31, wave = threadIdx.x >> 5;
  const int half = lane >> 4, lan = lane & 15;
  const size_t bh = (size_t)(b * NH + h) * (NSEQ * 64);
  const _Float16* Q = Qh + bh;
  const _Float16* K = Kh + bh;
  const _Float16* V = Vt + bh;                         // [d][n]
  const int nq0 = blockIdx.x * 128 + wave * 16;

  // Per-thread staging slots (one b128 per thread per tensor):
  // K chunk: 32 rows x 128B ; V chunk: 64 rows x 64B
  const int tid   = threadIdx.x;
  const int krow  = tid >> 3, kcol = (tid & 7) * 8;    // halves
  const int vrow  = tid >> 2, vcol = (tid & 3) * 8;    // halves
  unsigned skoff[2], svoff[2];
  skoff[0] = (unsigned)(uintptr_t)&sK[0][krow * 64 + kcol];
  skoff[1] = (unsigned)(uintptr_t)&sK[1][krow * 64 + kcol];
  svoff[0] = (unsigned)(uintptr_t)&sV[0][vrow * 32 + vcol];
  svoff[1] = (unsigned)(uintptr_t)&sV[1][vrow * 32 + vcol];

  // Q tile as two A fragments (d 0..31 and 32..63), pre-scaled by 1/8 in kernel 1
  H16 qa0, qa1;
  {
    const _Float16* qr = Q + (size_t)(nq0 + lan) * 64;
    qa0.s.lo = *(const v8h*)(qr + half * 16);
    qa0.s.hi = *(const v8h*)(qr + 8 + half * 16);
    qa1.s.lo = *(const v8h*)(qr + 32 + half * 16);
    qa1.s.hi = *(const v8h*)(qr + 40 + half * 16);
  }

  v8f oa[4] = {};                 // O accumulators, d-tiles 0..3 (cols d0=t*16)
  float mrow[8], lrow[8];
  #pragma unroll
  for (int r = 0; r < 8; ++r) { mrow[r] = -1e30f; lrow[r] = 0.0f; }

  // prologue: DMA chunk 0 into buffer 0
  async_copy_b128(skoff[0], K + (size_t)krow * 64 + kcol);
  async_copy_b128(svoff[0], V + (size_t)vrow * NSEQ + vcol);

  for (int it = 0; it < NSEQ / 32; ++it) {
    const int cur = it & 1, nxt = cur ^ 1;
    wait_async0();                 // this wave's DMA for chunk `it` has landed
    __syncthreads();               // everyone's DMA landed; prev buffer free

    if (it + 1 < NSEQ / 32) {      // prefetch chunk it+1 while computing
      const int m1 = (it + 1) * 32;
      async_copy_b128(skoff[nxt], K + (size_t)(m1 + krow) * 64 + kcol);
      async_copy_b128(svoff[nxt], V + (size_t)vrow * NSEQ + m1 + vcol);
    }

    // S = Q . K^T for two 16-column m-chunks (contraction over d = 64)
    const _Float16* kr1 = &sK[cur][lan * 64];
    const _Float16* kr2 = &sK[cur][(16 + lan) * 64];
    v16h b00 = *(const v16h*)(kr1 + half * 16);
    v16h b01 = *(const v16h*)(kr1 + 32 + half * 16);
    v16h b10 = *(const v16h*)(kr2 + half * 16);
    v16h b11 = *(const v16h*)(kr2 + 32 + half * 16);
    v8f s1 = {}, s2 = {};
    s1 = wmma_f16(qa0.v, b00, s1);
    s1 = wmma_f16(qa1.v, b01, s1);
    s2 = wmma_f16(qa0.v, b10, s2);
    s2 = wmma_f16(qa1.v, b11, s2);

    // online softmax: row stats live per lane-half (rows half*8 + r)
    float mx[8];
    #pragma unroll
    for (int r = 0; r < 8; ++r) mx[r] = fmaxf(s1[r], s2[r]);
    #pragma unroll
    for (int off = 1; off < 16; off <<= 1) {
      #pragma unroll
      for (int r = 0; r < 8; ++r) mx[r] = fmaxf(mx[r], __shfl_xor(mx[r], off, 32));
    }
    float al[8], rs[8];
    #pragma unroll
    for (int r = 0; r < 8; ++r) {
      float mn = fmaxf(mrow[r], mx[r]);
      al[r] = __expf(mrow[r] - mn);
      mrow[r] = mn;
      s1[r] = __expf(s1[r] - mn);
      s2[r] = __expf(s2[r] - mn);
      rs[r] = s1[r] + s2[r];
    }
    #pragma unroll
    for (int off = 1; off < 16; off <<= 1) {
      #pragma unroll
      for (int r = 0; r < 8; ++r) rs[r] += __shfl_xor(rs[r], off, 32);
    }
    #pragma unroll
    for (int r = 0; r < 8; ++r) lrow[r] = lrow[r] * al[r] + rs[r];
    #pragma unroll
    for (int t = 0; t < 4; ++t) {
      #pragma unroll
      for (int r = 0; r < 8; ++r) oa[t][r] *= al[r];
    }

    // transpose P (C layout -> A layout) through wave-private LDS
    _Float16* pb = &sP[wave][0];
    #pragma unroll
    for (int r = 0; r < 8; ++r) {
      pb[(half * 8 + r) * 32 + lan]      = (_Float16)s1[r];
      pb[(half * 8 + r) * 32 + 16 + lan] = (_Float16)s2[r];
    }
    asm volatile("s_wait_dscnt 0" ::: "memory");   // wave-private: split DS counter only
    H16 pa;
    pa.s.lo = *(const v8h*)(pb + lan * 32 + half * 16);
    pa.s.hi = *(const v8h*)(pb + lan * 32 + 8 + half * 16);

    // O += P . V  (contraction over m = 32), one WMMA per 16-wide d-tile
    #pragma unroll
    for (int t = 0; t < 4; ++t) {
      v16h vb = *(const v16h*)(&sV[cur][(t * 16 + lan) * 32 + half * 16]);
      oa[t] = wmma_f16(pa.v, vb, oa[t]);
    }
  }

  float inv[8];
  #pragma unroll
  for (int r = 0; r < 8; ++r) inv[r] = 1.0f / lrow[r];
  #pragma unroll
  for (int t = 0; t < 4; ++t) {
    v8h ov;
    #pragma unroll
    for (int r = 0; r < 8; ++r) ov[r] = (_Float16)(oa[t][r] * inv[r]);
    const int d = t * 16 + lan;
    const int c = d * 4 + h;                 // x.reshape(b, 64*4, n): c = d*H + h
    *(v8h*)(Xh + ((size_t)b * DMODEL + c) * NSEQ + nq0 + half * 8) = ov;
  }
}

// ---------------------------------------------------------------------------
// Kernel 3: output projection, f16 activations -> f32 result into d_out
// blockIdx: x = tile group (256), y = batch (4)
// ---------------------------------------------------------------------------
__global__ __launch_bounds__(256) void out_proj_kernel(
    const _Float16* __restrict__ Xh, const float* __restrict__ Wm,
    const float* __restrict__ bm, float* __restrict__ out)
{
  const int b    = blockIdx.y;
  const int lane = threadIdx.x & 31;
  const int wave = threadIdx.x >> 5;
  const int tile = blockIdx.x * 8 + wave;
  const int o0   = (tile & 15) * 16;
  const int n0   = (tile >> 4) * 16;
  const int half = lane >> 4, lan = lane & 15;
  const int n    = n0 + lan;

  const _Float16* Xb = Xh + (size_t)b * (DMODEL * NSEQ);
  const float* wr = Wm + (size_t)(o0 + lan) * DMODEL;

  v8f acc = {};
  #pragma unroll
  for (int k0 = 0; k0 < DMODEL; k0 += 32) {
    v16h a, bb;
    #pragma unroll
    for (int j = 0; j < 8; ++j) {
      a[j]     = (_Float16)wr[k0 + half * 16 + j];
      a[j + 8] = (_Float16)wr[k0 + half * 16 + 8 + j];
    }
    #pragma unroll
    for (int j = 0; j < 16; ++j)
      bb[j] = Xb[(size_t)(k0 + half * 16 + j) * NSEQ + n];
    acc = wmma_f16(a, bb, acc);
  }
  #pragma unroll
  for (int r = 0; r < 8; ++r)
    out[((size_t)b * DMODEL + o0 + half * 8 + r) * NSEQ + n] = acc[r] + bm[o0 + half * 8 + r];
}

// ---------------------------------------------------------------------------
extern "C" void kernel_launch(void* const* d_in, const int* in_sizes, int n_in,
                              void* d_out, int out_size, void* d_ws, size_t ws_size,
                              hipStream_t stream) {
  (void)in_sizes; (void)n_in; (void)out_size; (void)ws_size;
  const float* q_in = (const float*)d_in[0];
  const float* k_in = (const float*)d_in[1];
  const float* v_in = (const float*)d_in[2];
  const float* enc  = (const float*)d_in[3];
  const float* Wq   = (const float*)d_in[4];
  const float* bq   = (const float*)d_in[5];
  const float* Wk   = (const float*)d_in[6];
  const float* bk   = (const float*)d_in[7];
  const float* Wv   = (const float*)d_in[8];
  const float* bv   = (const float*)d_in[9];
  const float* Wm   = (const float*)d_in[10];
  const float* bm   = (const float*)d_in[11];

  const size_t tsz = (size_t)NB * NH * NSEQ * 64;   // 2M halves = 4 MiB each
  _Float16* Qh = (_Float16*)d_ws;
  _Float16* Kh = Qh + tsz;
  _Float16* Vt = Kh + tsz;                          // [b][h][64][2048]
  _Float16* Xh = Vt + tsz;                          // [b][256][2048], 4 MiB

  dim3 blk(256);
  qkv_proj_rope_kernel<<<dim3(256, NB, 3), blk, 0, stream>>>(
      q_in, k_in, v_in, Wq, Wk, Wv, bq, bk, bv, enc, Qh, Kh, Vt);
  attention_kernel<<<dim3(16, NH, NB), blk, 0, stream>>>(Qh, Kh, Vt, Xh);
  out_proj_kernel<<<dim3(256, NB), blk, 0, stream>>>(Xh, Wm, bm, (float*)d_out);
}
